// LRPEAttention_42039139893359
// MI455X (gfx1250) — matrix operand order repaired and checked by reading, not compiled, not run on hardware
//
#include <hip/hip_runtime.h>
#include <hip/hip_bf16.h>

typedef float    v2f  __attribute__((ext_vector_type(2)));
typedef float    v8f  __attribute__((ext_vector_type(8)));
typedef _Float16 v4h  __attribute__((ext_vector_type(4)));
typedef _Float16 v8h  __attribute__((ext_vector_type(8)));
typedef _Float16 v16h __attribute__((ext_vector_type(16)));

#define NTOK   2048
#define DMODEL 1024
#define NHEAD  16
#define HD     64
#define QKVLD  3072
#define WIN    99
#define NE     199   // 2*MAX_REL - 1

__device__ __forceinline__ v8f wmma4(v2f a, v2f b, v8f c) {
  // D = A(16x4,f32) * B(4x16,f32) + C
  return __builtin_amdgcn_wmma_f32_16x16x4_f32(
      false, a, false, b, (short)0, c, false, false);
}
__device__ __forceinline__ v8f wmma16(v16h a, v16h b, v8f c) {
  // D = A(16x32,f16) * B(32x16,f16) + C (f32 accumulate)
  return __builtin_amdgcn_wmma_f32_16x16x32_f16(
      false, a, false, b, (short)0, c, false, false);
}

__device__ __forceinline__ void cvt_split(const float4 x, v4h& hi, v4h& lo) {
  _Float16 h0 = (_Float16)x.x, h1 = (_Float16)x.y;
  _Float16 h2 = (_Float16)x.z, h3 = (_Float16)x.w;
  hi = (v4h){h0, h1, h2, h3};
  lo = (v4h){(_Float16)(x.x - (float)h0), (_Float16)(x.y - (float)h1),
             (_Float16)(x.z - (float)h2), (_Float16)(x.w - (float)h3)};
}

// Async global->LDS copy, 16B per lane, tracked by ASYNCcnt (CDNA5 TDM-lite).
// lds_off: wave-relative LDS byte address (low 32 bits of flat shared ptr).
__device__ __forceinline__ void async_copy_b128(unsigned lds_off, const void* g) {
  unsigned long long ga = (unsigned long long)(uintptr_t)g;
  asm volatile("global_load_async_to_lds_b128 %0, %1, off"
               :: "v"(lds_off), "v"(ga) : "memory");
}
__device__ __forceinline__ void wait_async0() {
  asm volatile("s_wait_asynccnt 0x0" ::: "memory");
}

// ---------------------------------------------------------------------------
// C[M,N] = A[M,K] @ B[N,K]^T + bias[N], fp32 in/out via split-f16 WMMA:
//   A*B ~= Ah*Bh + Ah*Bl + Al*Bh  (f32 accumulate, ~22-bit mantissa accuracy)
// block: 256 threads (8 waves), 64x64 C tile, K chunks of 32, pipelined stage.
// ---------------------------------------------------------------------------
__global__ __launch_bounds__(256) void gemm_nt_f16split_kernel(
    const float* __restrict__ A, const float* __restrict__ B,
    const float* __restrict__ bias, float* __restrict__ C,
    int M, int N, int K) {
  const int RS = 40;  // halves per LDS row: 80B stride -> conflict-free b128
  __shared__ _Float16 A_hi[64 * 40], A_lo[64 * 40];
  __shared__ _Float16 B_hi[64 * 40], B_lo[64 * 40];

  const int tid  = threadIdx.x;
  const int lane = tid & 31;
  const int w    = tid >> 5;
  const int half = lane >> 4;
  const int l16  = lane & 15;
  const int bm = blockIdx.y, bn = blockIdx.x;
  const int mi  = w & 3;
  const int ni0 = (w >> 2) * 2;

  // staging: this thread covers rows r0,r1 cols [c4,c4+4) of both 64x32 tiles
  const int r0 = tid >> 3, r1 = r0 + 32;
  const int c4 = (tid & 7) << 2;
  const float* Ap = A + (size_t)(bm * 64) * K + c4;
  const float* Bp = B + (size_t)(bn * 64) * K + c4;

  float4 ra0 = *(const float4*)(Ap + (size_t)r0 * K);
  float4 ra1 = *(const float4*)(Ap + (size_t)r1 * K);
  float4 rb0 = *(const float4*)(Bp + (size_t)r0 * K);
  float4 rb1 = *(const float4*)(Bp + (size_t)r1 * K);

  v8f acc0 = {0,0,0,0,0,0,0,0};
  v8f acc1 = {0,0,0,0,0,0,0,0};

  union U { v16h v; v8h p[2]; };

  for (int k0 = 0; k0 < K; k0 += 32) {
    v4h h, l;
    cvt_split(ra0, h, l); *(v4h*)&A_hi[r0*RS + c4] = h; *(v4h*)&A_lo[r0*RS + c4] = l;
    cvt_split(ra1, h, l); *(v4h*)&A_hi[r1*RS + c4] = h; *(v4h*)&A_lo[r1*RS + c4] = l;
    cvt_split(rb0, h, l); *(v4h*)&B_hi[r0*RS + c4] = h; *(v4h*)&B_lo[r0*RS + c4] = l;
    cvt_split(rb1, h, l); *(v4h*)&B_hi[r1*RS + c4] = h; *(v4h*)&B_lo[r1*RS + c4] = l;
    __syncthreads();

    if (k0 + 32 < K) {  // issue next chunk's global loads early (overlap WMMA)
      ra0 = *(const float4*)(Ap + (size_t)r0 * K + k0 + 32);
      ra1 = *(const float4*)(Ap + (size_t)r1 * K + k0 + 32);
      rb0 = *(const float4*)(Bp + (size_t)r0 * K + k0 + 32);
      rb1 = *(const float4*)(Bp + (size_t)r1 * K + k0 + 32);
    }

    // fragments: A lane k = half*16..+15 ; B lane k = half*16..+15
    const int arow = (mi * 16 + l16) * RS;
    U ah, al, bh0, bl0, bh1, bl1;
    ah.p[0] = *(const v8h*)&A_hi[arow + half * 16];
    ah.p[1] = *(const v8h*)&A_hi[arow + half * 16 + 8];
    al.p[0] = *(const v8h*)&A_lo[arow + half * 16];
    al.p[1] = *(const v8h*)&A_lo[arow + half * 16 + 8];
    const int brow0 = (ni0 * 16 + l16) * RS;
    const int brow1 = brow0 + 16 * RS;
    bh0.p[0] = *(const v8h*)&B_hi[brow0 + half * 16];
    bh0.p[1] = *(const v8h*)&B_hi[brow0 + half * 16 + 8];
    bl0.p[0] = *(const v8h*)&B_lo[brow0 + half * 16];
    bl0.p[1] = *(const v8h*)&B_lo[brow0 + half * 16 + 8];
    bh1.p[0] = *(const v8h*)&B_hi[brow1 + half * 16];
    bh1.p[1] = *(const v8h*)&B_hi[brow1 + half * 16 + 8];
    bl1.p[0] = *(const v8h*)&B_lo[brow1 + half * 16];
    bl1.p[1] = *(const v8h*)&B_lo[brow1 + half * 16 + 8];

    acc0 = wmma16(ah.v, bh0.v, acc0);
    acc0 = wmma16(ah.v, bl0.v, acc0);
    acc0 = wmma16(al.v, bh0.v, acc0);
    acc1 = wmma16(ah.v, bh1.v, acc1);
    acc1 = wmma16(ah.v, bl1.v, acc1);
    acc1 = wmma16(al.v, bh1.v, acc1);
    __syncthreads();
  }

  const int n0 = bn * 64 + ni0 * 16 + l16;
  const int n1 = n0 + 16;
  const float bv0 = bias[n0], bv1 = bias[n1];
  const int mbase = bm * 64 + mi * 16 + half * 8;
#pragma unroll
  for (int r = 0; r < 8; r++) {
    C[(size_t)(mbase + r) * N + n0] = acc0[r] + bv0;
    C[(size_t)(mbase + r) * N + n1] = acc1[r] + bv1;
  }
}

// ---------------------------------------------------------------------------
// Banded attention: pos = -1e8 outside |j-i|<=99 makes softmax exactly banded.
// One block per (16-query tile, head). 256 threads = 8 waves. Exact f32 WMMA.
// Q/K/rel staged via GLOBAL_LOAD_ASYNC_TO_LDS_B128 (ASYNCcnt).
// ---------------------------------------------------------------------------
__global__ __launch_bounds__(256) void attn_banded_kernel(
    const float* __restrict__ qkv, const float* __restrict__ rel,
    float* __restrict__ att) {
  __shared__ float q_s[16 * 68];     // 16 x 64, stride 68 (272B: 16B-aligned)
  __shared__ float k_s[240 * 68];    // up to 15 key tiles x 16 x 64
  __shared__ float rel_s[NE * 68];   // 199 x 64
  __shared__ float S_s[16 * 242];    // 16 x (<=240) logits, stride 242
  __shared__ float o_s[16 * 66];     // partial PV reduction
  const int SROW = 242;
  const int ROW  = 68;               // floats per row for q_s/k_s/rel_s

  const int tid  = threadIdx.x;
  const int lane = tid & 31;
  const int w    = tid >> 5;
  const int half = lane >> 4;
  const int l16  = lane & 15;

  const int qb = blockIdx.x;
  const int h  = blockIdx.y;
  const int i0 = qb * 16;

  int jlo = i0 - WIN;        if (jlo < 0) jlo = 0;
  int jhi = i0 + 15 + WIN;   if (jhi > NTOK - 1) jhi = NTOK - 1;
  const int t0 = jlo >> 4, t1 = jhi >> 4;
  const int ntiles = t1 - t0 + 1;        // 8..15
  const int jstart = t0 << 4;
  const int nk = ntiles << 4;

  const float scale = 0.125f;  // 1/sqrt(64)

  // ---- Phase 0: async global->LDS staging (float4 granularity) ---------
  {
    const unsigned qb_off = (unsigned)(uintptr_t)q_s;
    const unsigned kb_off = (unsigned)(uintptr_t)k_s;
    const unsigned rb_off = (unsigned)(uintptr_t)rel_s;
    {  // Q tile: exactly 256 float4
      int r = tid >> 4, c16 = tid & 15;
      async_copy_b128(qb_off + (r * ROW + c16 * 4) * 4,
                      qkv + (size_t)(i0 + r) * QKVLD + h * HD + c16 * 4);
    }
    for (int e = tid; e < nk * 16; e += 256) {  // K band
      int r = e >> 4, c16 = e & 15;
      async_copy_b128(kb_off + (r * ROW + c16 * 4) * 4,
                      qkv + (size_t)(jstart + r) * QKVLD + DMODEL + h * HD + c16 * 4);
    }
    for (int e = tid; e < NE * 16; e += 256) {  // rel_emb[h]
      int r = e >> 4, c16 = e & 15;
      async_copy_b128(rb_off + (r * ROW + c16 * 4) * 4,
                      rel + (size_t)h * NE * HD + r * HD + c16 * 4);
    }
    wait_async0();
  }
  __syncthreads();

  // ---- Phase A: S = scale * Q K^T per 16x16 key tile (WMMA f32) --------
  for (int t = w; t < ntiles; t += 8) {
    v8f acc = {0,0,0,0,0,0,0,0};
#pragma unroll
    for (int kk = 0; kk < HD; kk += 4) {
      v2f a = *(const v2f*)&q_s[l16 * ROW + kk + half * 2];
      v2f b = *(const v2f*)&k_s[(t * 16 + l16) * ROW + kk + half * 2];
      acc = wmma4(a, b, acc);
    }
#pragma unroll
    for (int r = 0; r < 8; r++)
      S_s[(r + half * 8) * SROW + t * 16 + l16] = acc[r] * scale;
  }
  __syncthreads();

  // ---- Phase A2: add relative-position logits (VALU), mask band --------
  {
    const int il = tid >> 4;
    const int jl = tid & 15;
    const float* qrow = &q_s[il * ROW];
    for (int t = 0; t < ntiles; t++) {
      int jg = jstart + t * 16 + jl;
      int delta = jg - (i0 + il) + WIN;
      float s;
      if (delta >= 0 && delta < NE) {
        const float* rrow = &rel_s[delta * ROW];
        float acc = 0.f;
#pragma unroll
        for (int d = 0; d < HD; d++) acc = fmaf(qrow[d], rrow[d], acc);
        s = S_s[il * SROW + t * 16 + jl] + acc;
      } else {
        s = -1.0e9f;   // exp underflows to exactly 0
      }
      S_s[il * SROW + t * 16 + jl] = s;
    }
  }
  __syncthreads();

  // ---- Phase B: row softmax (16 lanes per row, shuffle reductions) -----
  {
    const int il = tid >> 4;
    const int r  = tid & 15;
    float m = -3.0e38f;
    for (int js = r; js < nk; js += 16) m = fmaxf(m, S_s[il * SROW + js]);
#pragma unroll
    for (int msk = 8; msk >= 1; msk >>= 1) m = fmaxf(m, __shfl_xor(m, msk, 32));
    float sum = 0.f;
    for (int js = r; js < nk; js += 16) {
      float e = __expf(S_s[il * SROW + js] - m);
      S_s[il * SROW + js] = e;
      sum += e;
    }
#pragma unroll
    for (int msk = 8; msk >= 1; msk >>= 1) sum += __shfl_xor(sum, msk, 32);
    float inv = 1.0f / sum;
    for (int js = r; js < nk; js += 16) S_s[il * SROW + js] *= inv;
  }
  __syncthreads();

  // ---- Phase C: O = P V (WMMA f32), V streamed from global -------------
  {
    const int par = w >> 2;   // key-tile parity
    const int dch = w & 3;    // 16-wide d chunk
    v8f acc = {0,0,0,0,0,0,0,0};
    const float* vbase = qkv + 2 * DMODEL + h * HD + dch * 16 + l16;
    for (int t = par; t < ntiles; t += 2) {
#pragma unroll
      for (int kk = 0; kk < 4; kk++) {
        v2f a = *(const v2f*)&S_s[l16 * SROW + t * 16 + kk * 4 + half * 2];
        int jr = jstart + t * 16 + kk * 4 + half * 2;
        v2f b;
        b.x = vbase[(size_t)jr * QKVLD];
        b.y = vbase[(size_t)(jr + 1) * QKVLD];
        acc = wmma4(a, b, acc);
      }
    }
    if (par == 0) {
#pragma unroll
      for (int r = 0; r < 8; r++)
        o_s[(r + half * 8) * 66 + dch * 16 + l16] = acc[r];
    }
    __syncthreads();
    if (par == 1) {
#pragma unroll
      for (int r = 0; r < 8; r++) {
        int m = r + half * 8;
        float val = o_s[m * 66 + dch * 16 + l16] + acc[r];
        att[(size_t)(i0 + m) * DMODEL + h * HD + dch * 16 + l16] = val;
      }
    }
  }
}

// ---------------------------------------------------------------------------
extern "C" void kernel_launch(void* const* d_in, const int* in_sizes, int n_in,
                              void* d_out, int out_size, void* d_ws, size_t ws_size,
                              hipStream_t stream) {
  (void)in_sizes; (void)n_in; (void)out_size; (void)ws_size;
  const float* x       = (const float*)d_in[0];
  const float* qkv_w   = (const float*)d_in[1];
  const float* qkv_b   = (const float*)d_in[2];
  const float* proj_w  = (const float*)d_in[3];
  const float* proj_b  = (const float*)d_in[4];
  const float* rel_emb = (const float*)d_in[5];

  float* qkv = (float*)d_ws;                       // [2048, 3072]
  float* att = qkv + (size_t)NTOK * QKVLD;         // [2048, 1024]

  // 1) QKV projection: [2048,1024] @ [3072,1024]^T  (split-f16 WMMA)
  gemm_nt_f16split_kernel<<<dim3(QKVLD / 64, NTOK / 64), 256, 0, stream>>>(
      x, qkv_w, qkv_b, qkv, NTOK, QKVLD, DMODEL);

  // 2) Banded attention with learned relative positions (exact f32 WMMA)
  attn_banded_kernel<<<dim3(NTOK / 16, NHEAD), 256, 0, stream>>>(
      qkv, rel_emb, att);

  // 3) Output projection: [2048,1024] @ [1024,1024]^T (split-f16 WMMA)
  gemm_nt_f16split_kernel<<<dim3(DMODEL / 64, NTOK / 64), 256, 0, stream>>>(
      att, proj_w, proj_b, (float*)d_out, NTOK, DMODEL, DMODEL);
}
